// NeuronMemory_31035433681148
// MI455X (gfx1250) — matrix operand order repaired and checked by reading, not compiled, not run on hardware
//
#include <hip/hip_runtime.h>

// ---------------- problem constants ----------------
#define D_MODEL   1024
#define N_KNOW    32768
#define RANK      128
#define COARSE_K  20
#define FINE_K    10
#define NTOK      4096              // B*S = 2*2048
#define NSPLIT    8                 // router N-range splits
#define NRANGE    (N_KNOW / NSPLIT) // 4096 columns per block
#define NTILES    (NRANGE / 16)     // 256 16-col tiles per block

#define BSTRIDE   2064              // padded LDS row stride for B (bank-conflict free)
#define BBUF      (16 * BSTRIDE)    // 33024 B per B buffer
#define TILEB_OFF (2 * BBUF)        // 66048: per-wave 16x17 f32 scan tiles
#define SMEM_BYTES (TILEB_OFF + 8 * 1088)   // 74752 B dynamic LDS

#define NEG_INF   (-1e30f)

typedef float  v8f   __attribute__((ext_vector_type(8)));
typedef __bf16 v16bf __attribute__((ext_vector_type(16)));
typedef int    v4i   __attribute__((vector_size(16)));

#define AS1 __attribute__((address_space(1)))
#define AS3 __attribute__((address_space(3)))
// global ptr: drop const via plain cast, then addrspacecast generic->1
#define GCAST(p) ((AS1 v4i*)(v4i*)(p))
// lds ptr: addrspacecast generic->3
#define LCAST(p) ((AS3 v4i*)(p))

#if __has_builtin(__builtin_amdgcn_global_load_async_to_lds_b128)
#define HAVE_ASYNC_LDS 1
#else
#define HAVE_ASYNC_LDS 0
#endif

__device__ __forceinline__ void wait_asynccnt0() {
#if HAVE_ASYNC_LDS
#if __has_builtin(__builtin_amdgcn_s_wait_asynccnt)
  __builtin_amdgcn_s_wait_asynccnt(0);
#else
  __asm__ volatile("s_wait_asynccnt 0" ::: "memory");
#endif
#endif
}

__device__ __forceinline__ unsigned short f2bf(float f) {
  union { float f; unsigned u; } v; v.f = f;
  unsigned u = v.u;
  u += 0x7fffu + ((u >> 16) & 1u);      // round-to-nearest-even
  return (unsigned short)(u >> 16);
}
__device__ __forceinline__ unsigned pk2(float a, float b) {
  return (unsigned)f2bf(a) | ((unsigned)f2bf(b) << 16);
}

// =====================================================================
// Kernel 0: W_router [1024][32768] f32  ->  WT [32768][1024] bf16
// =====================================================================
__global__ __launch_bounds__(256) void k_transpose_bf16(
    const float* __restrict__ W, unsigned short* __restrict__ WT) {
  __shared__ float tile[32][33];
  const int nx = blockIdx.x * 32;   // n
  const int dy = blockIdx.y * 32;   // d
  const int tx = threadIdx.x, ty = threadIdx.y;  // (32,8)
  for (int j = ty; j < 32; j += 8)
    tile[j][tx] = W[(size_t)(dy + j) * N_KNOW + nx + tx];
  __syncthreads();
  for (int j = ty; j < 32; j += 8)
    WT[(size_t)(nx + j) * D_MODEL + dy + tx] = f2bf(tile[tx][j]);
}

// =====================================================================
// Kernel 1: fused router GEMM (bf16 WMMA) + per-(token, n-split) top-20
//   grid = (NSPLIT, NTOK/128), block = 256 threads = 8 waves
//   wave w owns tokens tokbase + w*16..+15; full K=1024 A in registers.
//   B tiles double-buffered in LDS; staged async (ASYNCcnt) when available.
// =====================================================================
__global__ __launch_bounds__(256, 1) void k_router_topk(
    const float* __restrict__ x, const unsigned short* __restrict__ WT,
    float* __restrict__ Ps, int* __restrict__ Pi) {
  extern __shared__ __align__(16) char smem[];
  float* tileb  = (float*)(smem + TILEB_OFF);
  float* cand_s = (float*)smem;            // reused after final barrier
  int*   cand_i = (int*)(smem + 20480);

  const int t    = threadIdx.x;
  const int lane = t & 31, wid = t >> 5;
  const int hl   = lane >> 4, ln = lane & 15;
  const int tokbase = blockIdx.y * 128;
  const int nbase   = blockIdx.x * NRANGE;

  // ---- A matrix (16 tokens x K=1024 bf16) resident in registers ----
  // A 16x32 k order used here (contiguous, same as B): lanes 0-15 hold K {0..15}, lanes 16-31 {16..31}
  v16bf areg[32];
  {
    const float* xr = x + (size_t)(tokbase + wid * 16 + ln) * D_MODEL;
    const float4* x4 = (const float4*)xr;
#pragma unroll
    for (int s = 0; s < 32; ++s) {
      const int k0 = (s * 32 + hl * 16) >> 2;       // in float4 units
      const int k1 = (s * 32 + hl * 16 + 8) >> 2;
      float4 c0 = x4[k0], c1 = x4[k0 + 1];
      float4 c2 = x4[k1], c3 = x4[k1 + 1];
      union { v16bf v; unsigned u[8]; } a;
      a.u[0] = pk2(c0.x, c0.y); a.u[1] = pk2(c0.z, c0.w);
      a.u[2] = pk2(c1.x, c1.y); a.u[3] = pk2(c1.z, c1.w);
      a.u[4] = pk2(c2.x, c2.y); a.u[5] = pk2(c2.z, c2.w);
      a.u[6] = pk2(c3.x, c3.y); a.u[7] = pk2(c3.z, c3.w);
      areg[s] = a.v;
    }
  }

  // per-lane sorted (desc) top-20 over the columns this lane scans
  float ts[20]; int ti[20];
#pragma unroll
  for (int j = 0; j < 20; ++j) { ts[j] = NEG_INF; ti[j] = 0; }

  float* myTile = tileb + wid * 272;   // 16 x 17 f32

  // ---- prologue: stage tile 0 into buffer 0 ----
  {
    const uint4* src = (const uint4*)(WT + (size_t)nbase * D_MODEL);
#pragma unroll
    for (int it = 0; it < 8; ++it) {
      int e = t + it * 256;
      int r = e >> 7, c = e & 127;
#if HAVE_ASYNC_LDS
      __builtin_amdgcn_global_load_async_to_lds_b128(
          GCAST(src + (size_t)r * 128 + c),
          LCAST(smem + r * BSTRIDE + c * 16), 0, 0);
#else
      *(uint4*)(smem + r * BSTRIDE + c * 16) = src[(size_t)r * 128 + c];
#endif
    }
    wait_asynccnt0();
  }
  __syncthreads();

  for (int nt = 0; nt < NTILES; ++nt) {
    const int n0 = nbase + nt * 16;
    char* cur = smem + (nt & 1) * BBUF;
    char* nxt = smem + ((nt + 1) & 1) * BBUF;

    // ---- issue stage of tile nt+1 (overlaps with compute below) ----
#if !HAVE_ASYNC_LDS
    uint4 sreg[8];
#endif
    if (nt + 1 < NTILES) {
      const uint4* src = (const uint4*)(WT + (size_t)(n0 + 16) * D_MODEL);
#pragma unroll
      for (int it = 0; it < 8; ++it) {
        int e = t + it * 256;
        int r = e >> 7, c = e & 127;
#if HAVE_ASYNC_LDS
        __builtin_amdgcn_global_load_async_to_lds_b128(
            GCAST(src + (size_t)r * 128 + c),
            LCAST(nxt + r * BSTRIDE + c * 16), 0, 0);
#else
        sreg[it] = src[(size_t)r * 128 + c];
#endif
      }
    }

    // ---- 16x16 logits tile, K=1024: 32 WMMAs, 4 independent acc chains,
    //      B reads software-pipelined 2 steps ahead ----
    v8f acc0, acc1, acc2, acc3;
#pragma unroll
    for (int j = 0; j < 8; ++j) { acc0[j] = 0.f; acc1[j] = 0.f; acc2[j] = 0.f; acc3[j] = 0.f; }
    const char* bbase = cur + ln * BSTRIDE + hl * 32;
    v16bf bb[2];
    bb[0] = *(const v16bf*)(bbase);
    bb[1] = *(const v16bf*)(bbase + 64);
#pragma unroll
    for (int s = 0; s < 32; ++s) {
      v16bf bcur = bb[s & 1];
      if (s + 2 < 32) bb[s & 1] = *(const v16bf*)(bbase + (s + 2) * 64);
      if ((s & 3) == 0)
        acc0 = __builtin_amdgcn_wmma_f32_16x16x32_bf16(false, areg[s], false, bcur, (short)0, acc0, false, false);
      else if ((s & 3) == 1)
        acc1 = __builtin_amdgcn_wmma_f32_16x16x32_bf16(false, areg[s], false, bcur, (short)0, acc1, false, false);
      else if ((s & 3) == 2)
        acc2 = __builtin_amdgcn_wmma_f32_16x16x32_bf16(false, areg[s], false, bcur, (short)0, acc2, false, false);
      else
        acc3 = __builtin_amdgcn_wmma_f32_16x16x32_bf16(false, areg[s], false, bcur, (short)0, acc3, false, false);
    }
    v8f acc = (acc0 + acc1) + (acc2 + acc3);

    // ---- in-wave transpose via LDS tile, then register top-20 insert ----
#pragma unroll
    for (int r = 0; r < 8; ++r)
      myTile[(r + 8 * hl) * 17 + ln] = acc[r];   // C layout: row r+8*hl, col ln
    __asm__ volatile("" ::: "memory");           // LDS is in-order per wave
#pragma unroll
    for (int j = 0; j < 8; ++j) {
      float vv = myTile[ln * 17 + hl * 8 + j];   // token = wid*16+ln
      if (vv > ts[19]) {
        float cv = vv; int cidx = n0 + hl * 8 + j;
#pragma unroll
        for (int p = 0; p < 20; ++p) {
          bool gt = cv > ts[p];
          float os = ts[p]; int oi = ti[p];
          ts[p] = gt ? cv : os;  ti[p] = gt ? cidx : oi;
          cv    = gt ? os : cv;  cidx  = gt ? oi : cidx;
        }
      }
    }

    // ---- finish staging of nt+1, then one barrier per tile ----
#if !HAVE_ASYNC_LDS
    if (nt + 1 < NTILES) {
#pragma unroll
      for (int it = 0; it < 8; ++it) {
        int e = t + it * 256;
        int r = e >> 7, c = e & 127;
        *(uint4*)(nxt + r * BSTRIDE + c * 16) = sreg[it];
      }
    }
#else
    wait_asynccnt0();
#endif
    (void)nxt;
    __syncthreads();
  }

  // ---- dump per-lane lists (LDS reused; all waves past final barrier) ----
  {
    int cb = (wid * 32 + lane) * 20;
#pragma unroll
    for (int j = 0; j < 20; ++j) { cand_s[cb + j] = ts[j]; cand_i[cb + j] = ti[j]; }
  }
  __syncthreads();

  // merge the two half-lane lists of each token -> sorted top-20 partial
  if (t < 128) {
    const int w = t >> 4, m = t & 15;
    const int b0 = (w * 32 + m) * 20, b1 = (w * 32 + 16 + m) * 20;
    int p0 = 0, p1 = 0;
    const size_t ob = ((size_t)(tokbase + t) * NSPLIT + blockIdx.x) * 20;
    for (int r = 0; r < 20; ++r) {
      float v0 = (p0 < 20) ? cand_s[b0 + p0] : NEG_INF;
      float v1 = (p1 < 20) ? cand_s[b1 + p1] : NEG_INF;
      bool take0 = v0 >= v1;
      float bv = take0 ? v0 : v1;
      int   bi = take0 ? cand_i[b0 + p0] : cand_i[b1 + p1];
      if (take0) p0++; else p1++;
      Ps[ob + r] = bv; Pi[ob + r] = bi;
    }
  }
}

// =====================================================================
// Kernel 2: per-token merge -> coarse top-20, query, fine top-10,
//           softmax, weighted V gather, all outputs. grid = NTOK blocks.
// =====================================================================
__global__ __launch_bounds__(256) void k_finalize(
    const float* __restrict__ x, const float* __restrict__ W_enc,
    const float* __restrict__ K_all, const float* __restrict__ V_all,
    const float* __restrict__ Ps, const int* __restrict__ Pi,
    float* __restrict__ out, float* __restrict__ o_cidx,
    float* __restrict__ o_cscore, float* __restrict__ o_fidx,
    float* __restrict__ o_fw) {
  const int tok = blockIdx.x;
  const int t = threadIdx.x;

  __shared__ float cs[COARSE_K]; __shared__ int ci[COARSE_K];
  __shared__ float q[RANK];
  __shared__ float qp[256];
  __shared__ float fsc[COARSE_K][8];
  __shared__ float fw[FINE_K]; __shared__ int fgi[FINE_K];

  // ---- merge NSPLIT sorted partial lists via wave-wide argmax ----
  if (t < 32) {
    int pos = 0;
    const size_t lb = ((size_t)tok * NSPLIT + t) * 20;   // valid for t < NSPLIT
    for (int r = 0; r < COARSE_K; ++r) {
      float hv = (t < NSPLIT && pos < 20) ? Ps[lb + pos] : NEG_INF;
      float bv = hv; int bl = t;
      for (int off = 16; off; off >>= 1) {
        float ov = __shfl_xor(bv, off, 32);
        int   ol = __shfl_xor(bl, off, 32);
        if (ov > bv || (ov == bv && ol < bl)) { bv = ov; bl = ol; }
      }
      if (t == bl && t < NSPLIT) { cs[r] = bv; ci[r] = Pi[lb + pos]; pos++; }
    }
  }
  __syncthreads();

  // ---- query = x[tok] @ W_enc  (two-way split over d, LDS reduce) ----
  {
    const int r = t & 127, h = t >> 7;
    const float* xr = x + (size_t)tok * D_MODEL;
    float s = 0.f;
    for (int d = h * 512; d < h * 512 + 512; ++d)
      s += xr[d] * W_enc[(size_t)d * RANK + r];
    qp[t] = s;
  }
  __syncthreads();
  if (t < RANK) q[t] = qp[t] + qp[t + 128];
  if (t < COARSE_K) { o_cidx[(size_t)tok * COARSE_K + t] = (float)ci[t];
                      o_cscore[(size_t)tok * COARSE_K + t] = cs[t]; }
  __syncthreads();

  // ---- fine scores: 20 candidates x (8 threads x 16 dims) ----
  if (t < COARSE_K * 8) {
    const int c = t >> 3, j = t & 7;
    const float* kr = K_all + (size_t)ci[c] * RANK + j * 16;
    float s = 0.f;
#pragma unroll
    for (int e = 0; e < 16; ++e) s += kr[e] * q[j * 16 + e];
    fsc[c][j] = s;
  }
  __syncthreads();

  // ---- reduce, fine top-10, softmax (thread 0; tiny) ----
  if (t == 0) {
    float fs[COARSE_K];
    for (int c = 0; c < COARSE_K; ++c) {
      float a = 0.f;
      for (int j = 0; j < 8; ++j) a += fsc[c][j];
      fs[c] = a * 0.08838834764831845f;   // 1/sqrt(128)
    }
    float wv[FINE_K]; int wi[FINE_K]; bool used[COARSE_K];
    for (int c = 0; c < COARSE_K; ++c) used[c] = false;
    for (int r = 0; r < FINE_K; ++r) {
      float b = NEG_INF; int bi = 0;
      for (int c = 0; c < COARSE_K; ++c)
        if (!used[c] && fs[c] > b) { b = fs[c]; bi = c; }
      used[bi] = true; wv[r] = b; wi[r] = bi;
    }
    float mx = wv[0], sum = 0.f;
    for (int r = 0; r < FINE_K; ++r) { wv[r] = __expf(wv[r] - mx); sum += wv[r]; }
    float inv = 1.f / sum;
    for (int r = 0; r < FINE_K; ++r) {
      fw[r] = wv[r] * inv; fgi[r] = ci[wi[r]];
      o_fidx[(size_t)tok * FINE_K + r] = (float)fgi[r];
      o_fw[(size_t)tok * FINE_K + r]   = fw[r];
    }
  }
  __syncthreads();

  // ---- output = sum_k fw[k] * V_all[fgi[k]]  (coalesced float4) ----
  {
    const int d = t * 4;
    float4 a; a.x = 0.f; a.y = 0.f; a.z = 0.f; a.w = 0.f;
#pragma unroll
    for (int k = 0; k < FINE_K; ++k) {
      const float4 v = *(const float4*)(V_all + (size_t)fgi[k] * D_MODEL + d);
      const float w = fw[k];
      a.x += w * v.x; a.y += w * v.y; a.z += w * v.z; a.w += w * v.w;
    }
    *(float4*)(out + (size_t)tok * D_MODEL + d) = a;
  }
}

// =====================================================================
extern "C" void kernel_launch(void* const* d_in, const int* in_sizes, int n_in,
                              void* d_out, int out_size, void* d_ws, size_t ws_size,
                              hipStream_t stream) {
  (void)in_sizes; (void)n_in; (void)out_size; (void)ws_size;
  const float* x        = (const float*)d_in[0];
  const float* W_router = (const float*)d_in[1];
  const float* W_enc    = (const float*)d_in[2];
  const float* K_all    = (const float*)d_in[3];
  const float* V_all    = (const float*)d_in[4];
  float* out = (float*)d_out;

  // workspace layout
  unsigned short* WT = (unsigned short*)d_ws;                         // 64 MB
  const size_t WT_BYTES = (size_t)N_KNOW * D_MODEL * 2;
  float* Ps = (float*)((char*)d_ws + WT_BYTES);                       // 2.5 MB
  const size_t PS_BYTES = (size_t)NTOK * NSPLIT * 20 * 4;
  int*   Pi = (int*)((char*)d_ws + WT_BYTES + PS_BYTES);              // 2.5 MB

  // flattened outputs: output, coarse_indices, coarse_scores, fine_indices, fine_weights
  float* o_out    = out;
  float* o_cidx   = out + (size_t)NTOK * D_MODEL;
  float* o_cscore = o_cidx + (size_t)NTOK * COARSE_K;
  float* o_fidx   = o_cscore + (size_t)NTOK * COARSE_K;
  float* o_fw     = o_fidx + (size_t)NTOK * FINE_K;

  // opt into >64KB dynamic LDS for the router kernel (CDNA5: 320KB/WGP)
  (void)hipFuncSetAttribute(reinterpret_cast<const void*>(k_router_topk),
                            hipFuncAttributeMaxDynamicSharedMemorySize, SMEM_BYTES);

  k_transpose_bf16<<<dim3(N_KNOW / 32, D_MODEL / 32), dim3(32, 8), 0, stream>>>(
      W_router, WT);
  k_router_topk<<<dim3(NSPLIT, NTOK / 128), 256, SMEM_BYTES, stream>>>(x, WT, Ps, Pi);
  k_finalize<<<dim3(NTOK), 256, 0, stream>>>(x, W_enc, K_all, V_all, Ps, Pi,
                                             o_out, o_cidx, o_cscore, o_fidx, o_fw);
}
